// SSD_53060025974995
// MI455X (gfx1250) — matrix operand order repaired and checked by reading, not compiled, not run on hardware
//
#include <hip/hip_runtime.h>
#include <hip/hip_bf16.h>

typedef __attribute__((ext_vector_type(16))) _Float16 v16h;
typedef __attribute__((ext_vector_type(8)))  float    v8f;

#define BM 64
#define BN 128
#define KT 32
#define LDK 40   // 32 + 8 halves pad: 80B row stride, 16B-aligned, bank-conflict friendly

union FragU { v16h h; uint4 q[2]; };

__device__ inline v16h ld2q(const _Float16* p0, const _Float16* p1) {
  FragU u;
  u.q[0] = *(const uint4*)p0;
  u.q[1] = *(const uint4*)p1;
  return u.h;
}

// Implicit-GEMM convolution: D[m][n] = sum_k W[m][k] * im2col(X)[k][n] + bias[m]
// m = cout, n = (b, oh, ow), k = (ci, kh, kw).
// Activations are f16 (NCHW), weights f32 (converted while staging), f32 accumulate.
// Double-buffered LDS: stage K-slab kt+KT while issuing WMMAs on slab kt.
__global__ __launch_bounds__(256)
void conv_wmma_kernel(const _Float16* __restrict__ X, const float* __restrict__ Wt,
                      const float* __restrict__ bias,
                      _Float16* __restrict__ OutH,   // layout 0: NCHW f16 intermediate
                      float* __restrict__ OutF,      // layout 1: channel-last f32 concat
                      int Bn, int Cin, int H, int Win, int Cout, int Ho, int Wo,
                      int ks, int st, int pd, int dl, int doRelu,
                      int layout, int obstride, int obase)
{
  __shared__ __align__(16) _Float16 lA[2][BM * LDK];
  __shared__ __align__(16) _Float16 lB[2][BN * LDK];

  const int tid  = threadIdx.x;
  const int lane = tid & 31;
  const int wid  = tid >> 5;
  const int waveM = wid & 1;        // 2 waves along M
  const int waveN = wid >> 1;       // 4 waves along N
  const int lmod = lane & 15;
  const int laneHalf = lane >> 4;

  const int HW   = Ho * Wo;
  const int Ntot = Bn * HW;
  const int K    = Cin * ks * ks;
  const int nBlock = blockIdx.x * BN;
  const int mBlock = blockIdx.y * BM;

  v8f c00 = {}, c01 = {}, c10 = {}, c11 = {};

  // B-tile (im2col) fill mapping: 2 threads per column, 16 consecutive k each
  const int nL  = tid >> 1;
  const int kLB = (tid & 1) * 16;
  const int nG  = nBlock + nL;
  const bool nValid = nG < Ntot;
  int bB = 0, ohB = 0, owB = 0;
  if (nValid) { bB = nG / HW; int r = nG - bB * HW; ohB = r / Wo; owB = r - ohB * Wo; }
  const int ihBase = ohB * st - pd;
  const int iwBase = owB * st - pd;
  const _Float16* Xb = X + (size_t)bB * Cin * H * Win;

  // A-tile (weights) fill mapping: 4 threads per row, 8 consecutive k each
  const int mL  = tid >> 2;
  const int kLA = (tid & 3) * 8;
  const int mG  = mBlock + mL;

  auto stage = [&](int bi, int kt) {
    // --- weights: f32 -> f16 ---
    {
      const float* wrow = Wt + (size_t)mG * K + (kt + kLA);
      #pragma unroll
      for (int i = 0; i < 8; ++i) {
        int kG = kt + kLA + i;
        float v = (mG < Cout && kG < K) ? wrow[i] : 0.0f;
        lA[bi][mL * LDK + kLA + i] = (_Float16)v;
      }
    }
    // --- activations: f16 copy with carry-incremented (ci,kh,kw) ---
    {
      const int kG0 = kt + kLB;
      int ci, kh, kw;
      if (ks == 1) { ci = kG0; kh = 0; kw = 0; }
      else { ci = kG0 / 9; int rr = kG0 - ci * 9; kh = rr / 3; kw = rr - kh * 3; }
      #pragma unroll
      for (int i = 0; i < 16; ++i) {
        _Float16 v = (_Float16)0.0f;
        if (nValid && (kG0 + i) < K) {
          int ih = ihBase + kh * dl;
          int iw = iwBase + kw * dl;
          if (ih >= 0 && ih < H && iw >= 0 && iw < Win)
            v = Xb[((size_t)ci * H + ih) * Win + iw];
        }
        lB[bi][nL * LDK + kLB + i] = v;
        if (ks == 1) { ++ci; }
        else if (++kw == 3) { kw = 0; if (++kh == 3) { kh = 0; ++ci; } }
      }
    }
  };

  auto compute = [&](int bi) {
    // A fragment: lane holds row m; K pairs 0-7/16-23 (lanes<16) vs 8-15/24-31
    const _Float16* ra0 = &lA[bi][(waveM * 32 +      lmod) * LDK + laneHalf * 8];
    const _Float16* ra1 = &lA[bi][(waveM * 32 + 16 + lmod) * LDK + laneHalf * 8];
    v16h a0 = ld2q(ra0, ra0 + 16);
    v16h a1 = ld2q(ra1, ra1 + 16);
    // B fragment: lane holds col n; K 0-7/16-23 (lanes<16) vs 8-15/24-31
    const _Float16* rb0 = &lB[bi][(waveN * 32 +      lmod) * LDK + laneHalf * 8];
    const _Float16* rb1 = &lB[bi][(waveN * 32 + 16 + lmod) * LDK + laneHalf * 8];
    v16h b0 = ld2q(rb0, rb0 + 16);
    v16h b1 = ld2q(rb1, rb1 + 16);

    c00 = __builtin_amdgcn_wmma_f32_16x16x32_f16(false, a0, false, b0, (short)0, c00, false, false);
    c01 = __builtin_amdgcn_wmma_f32_16x16x32_f16(false, a0, false, b1, (short)0, c01, false, false);
    c10 = __builtin_amdgcn_wmma_f32_16x16x32_f16(false, a1, false, b0, (short)0, c10, false, false);
    c11 = __builtin_amdgcn_wmma_f32_16x16x32_f16(false, a1, false, b1, (short)0, c11, false, false);
  };

  stage(0, 0);
  __syncthreads();
  int bi = 0;
  for (int kt = 0; kt < K; kt += KT) {
    if (kt + KT < K) {
      stage(bi ^ 1, kt + KT);                       // overlap next-slab VMEM with WMMA
      if (mG < Cout && kt + 2 * KT < K)             // CDNA5 global_prefetch_b8 for K+2
        __builtin_prefetch(Wt + (size_t)mG * K + (kt + 2 * KT + kLA), 0, 3);
    }
    compute(bi);
    __syncthreads();
    bi ^= 1;
  }

  // --- epilogue: bias + ReLU; NCHW f16 (layout 0) or channel-last f32 concat (layout 1)
  const int mW = mBlock + waveM * 32;
  const int nW = nBlock + waveN * 32;
  auto storeTile = [&](const v8f& cc, int m0, int n0) {
    int n = n0 + lmod;
    if (n >= Ntot) return;
    int bb = n / HW; int r = n - bb * HW; int oh = r / Wo; int ow = r - oh * Wo;
    #pragma unroll
    for (int rr2 = 0; rr2 < 8; ++rr2) {
      int m = m0 + rr2 + laneHalf * 8;   // C/D layout: row = vgpr + (lane>=16 ? 8 : 0)
      if (m >= Cout) continue;
      float v = cc[rr2] + bias[m];
      if (doRelu) v = fmaxf(v, 0.0f);
      if (layout == 0)
        OutH[(((size_t)bb * Cout + m) * Ho + oh) * Wo + ow] = (_Float16)v;
      else
        OutF[(size_t)bb * obstride + obase + (size_t)(oh * Wo + ow) * Cout + m] = v;
    }
  };
  storeTile(c00, mW, nW);
  storeTile(c01, mW, nW + 16);
  storeTile(c10, mW + 16, nW);
  storeTile(c11, mW + 16, nW + 16);
}

__global__ void f32_to_f16_kernel(const float* __restrict__ in, _Float16* __restrict__ out, int n)
{
  int idx = blockIdx.x * blockDim.x + threadIdx.x;
  if (idx < n) out[idx] = (_Float16)in[idx];
}

__global__ void maxpool_kernel(const _Float16* __restrict__ in, _Float16* __restrict__ out,
                               int Bn, int C, int H, int W, int Ho, int Wo,
                               int k, int s, int pt, int pl)
{
  int idx = blockIdx.x * blockDim.x + threadIdx.x;
  int tot = Bn * C * Ho * Wo;
  if (idx >= tot) return;
  int ow = idx % Wo; int t = idx / Wo;
  int oh = t % Ho; t /= Ho;
  int c = t % C; int b = t / C;
  float m = -3.402823466e38f;
  for (int i = 0; i < k; ++i)
    for (int j = 0; j < k; ++j) {
      int ih = oh * s - pt + i, iw = ow * s - pl + j;
      if (ih >= 0 && ih < H && iw >= 0 && iw < W)
        m = fmaxf(m, (float)in[(((size_t)b * C + c) * H + ih) * W + iw]);
    }
  out[idx] = (_Float16)m;
}

__global__ void l2norm_kernel(const _Float16* __restrict__ in, const float* __restrict__ wt,
                              _Float16* __restrict__ out, int Bn, int C, int HW)
{
  int idx = blockIdx.x * blockDim.x + threadIdx.x;
  if (idx >= Bn * HW) return;
  int b = idx / HW, p = idx - b * HW;
  const _Float16* base = in + (size_t)b * C * HW + p;
  float s = 0.0f;
  for (int c = 0; c < C; ++c) { float v = (float)base[(size_t)c * HW]; s += v * v; }
  float inv = 1.0f / (sqrtf(s) + 1e-10f);
  _Float16* obase = out + (size_t)b * C * HW + p;
  for (int c = 0; c < C; ++c)
    obase[(size_t)c * HW] = (_Float16)(wt[c] * (float)base[(size_t)c * HW] * inv);
}

__global__ void copy_kernel(const float* __restrict__ in, float* __restrict__ out, int n)
{
  int idx = blockIdx.x * blockDim.x + threadIdx.x;
  if (idx < n) out[idx] = in[idx];
}

extern "C" void kernel_launch(void* const* d_in, const int* in_sizes, int n_in,
                              void* d_out, int out_size, void* d_ws, size_t ws_size,
                              hipStream_t stream)
{
  (void)in_sizes; (void)n_in; (void)out_size; (void)ws_size;
  // Input index map (setup_inputs dict order, pytree-flattened):
  // 0: x | 1..30: vgg (W,b)*15 | 31..46: extras (W,b)*8 | 47..58: loc (W,b)*6
  // 59..70: conf (W,b)*6 | 71: l2_weight | 72: dbox_list
  const float* x = (const float*)d_in[0];
  auto Wp = [&](int i) { return (const float*)d_in[i]; };
  const int VGG = 1, EXT = 31, LOC = 47, CONF = 59;
  const float* l2w  = (const float*)d_in[71];
  const float* dbox = (const float*)d_in[72];

  _Float16* ws = (_Float16*)d_ws;
  size_t off = 0;
  auto carve = [&](size_t n) { _Float16* p = ws + off; off += (n + 63) & ~(size_t)63; return p; };
  _Float16* X16  = carve(4ull * 3 * 300 * 300);
  _Float16* bufA = carve(4ull * 64 * 300 * 300);
  _Float16* bufB = carve(4ull * 64 * 300 * 300);
  _Float16* S0 = carve(4ull * 512 * 38 * 38);
  _Float16* S1 = carve(4ull * 1024 * 19 * 19);
  _Float16* S2 = carve(4ull * 512 * 10 * 10);
  _Float16* S3 = carve(4ull * 256 * 5 * 5);
  _Float16* S4 = carve(4ull * 256 * 3 * 3);
  _Float16* S5 = carve(4ull * 256 * 1 * 1);

  auto convN = [&](const _Float16* Xp, int wi, _Float16* Op,
                   int Cin, int H, int Wd, int Cout, int Ho, int Wo,
                   int ks, int st, int pd, int dl) {
    dim3 g((4 * Ho * Wo + BN - 1) / BN, (Cout + BM - 1) / BM);
    conv_wmma_kernel<<<g, 256, 0, stream>>>(Xp, Wp(wi), Wp(wi + 1), Op, nullptr,
        4, Cin, H, Wd, Cout, Ho, Wo, ks, st, pd, dl, /*relu*/1, /*layout*/0, 0, 0);
  };
  auto convHead = [&](const _Float16* Xp, int wi, float* Op,
                      int Cin, int H, int Wd, int Cout, int obst, int obas) {
    dim3 g((4 * H * Wd + BN - 1) / BN, (Cout + BM - 1) / BM);
    conv_wmma_kernel<<<g, 256, 0, stream>>>(Xp, Wp(wi), Wp(wi + 1), nullptr, Op,
        4, Cin, H, Wd, Cout, H, Wd, 3, 1, 1, 1, /*relu*/0, /*layout*/1, obst, obas);
  };
  auto pool = [&](const _Float16* Xp, _Float16* Op, int C, int H, int Wd,
                  int Ho, int Wo, int k, int st, int pt, int pl) {
    int tot = 4 * C * Ho * Wo;
    maxpool_kernel<<<(tot + 255) / 256, 256, 0, stream>>>(Xp, Op, 4, C, H, Wd, Ho, Wo, k, st, pt, pl);
  };

  // input fp32 -> fp16 once
  {
    int n = 4 * 3 * 300 * 300;
    f32_to_f16_kernel<<<(n + 255) / 256, 256, 0, stream>>>(x, X16, n);
  }

  // --- VGG backbone ---
  convN(X16,  VGG + 0,  bufA, 3,  300, 300, 64,  300, 300, 3, 1, 1, 1);
  convN(bufA, VGG + 2,  bufB, 64, 300, 300, 64,  300, 300, 3, 1, 1, 1);
  pool(bufB, bufA, 64, 300, 300, 150, 150, 2, 2, 0, 0);
  convN(bufA, VGG + 4,  bufB, 64, 150, 150, 128, 150, 150, 3, 1, 1, 1);
  convN(bufB, VGG + 6,  bufA, 128,150, 150, 128, 150, 150, 3, 1, 1, 1);
  pool(bufA, bufB, 128, 150, 150, 75, 75, 2, 2, 0, 0);
  convN(bufB, VGG + 8,  bufA, 128, 75, 75, 256, 75, 75, 3, 1, 1, 1);
  convN(bufA, VGG + 10, bufB, 256, 75, 75, 256, 75, 75, 3, 1, 1, 1);
  convN(bufB, VGG + 12, bufA, 256, 75, 75, 256, 75, 75, 3, 1, 1, 1);
  pool(bufA, bufB, 256, 75, 75, 38, 38, 2, 2, 0, 0);          // ceil-mode via OOB skip
  convN(bufB, VGG + 14, bufA, 256, 38, 38, 512, 38, 38, 3, 1, 1, 1);
  convN(bufA, VGG + 16, bufB, 512, 38, 38, 512, 38, 38, 3, 1, 1, 1);
  convN(bufB, VGG + 18, bufA, 512, 38, 38, 512, 38, 38, 3, 1, 1, 1);
  l2norm_kernel<<<(4 * 38 * 38 + 255) / 256, 256, 0, stream>>>(bufA, l2w, S0, 4, 512, 38 * 38);
  pool(bufA, bufB, 512, 38, 38, 19, 19, 2, 2, 0, 0);
  convN(bufB, VGG + 20, bufA, 512, 19, 19, 512, 19, 19, 3, 1, 1, 1);
  convN(bufA, VGG + 22, bufB, 512, 19, 19, 512, 19, 19, 3, 1, 1, 1);
  convN(bufB, VGG + 24, bufA, 512, 19, 19, 512, 19, 19, 3, 1, 1, 1);
  pool(bufA, bufB, 512, 19, 19, 19, 19, 3, 1, 1, 1);          // pool5 k3 s1 p1
  convN(bufB, VGG + 26, bufA, 512, 19, 19, 1024, 19, 19, 3, 1, 6, 6);   // conv6 dilated
  convN(bufA, VGG + 28, S1,   1024,19, 19, 1024, 19, 19, 1, 1, 0, 1);   // conv7 -> source1

  // --- extras ---
  convN(S1,   EXT + 0,  bufA, 1024,19, 19, 256, 19, 19, 1, 1, 0, 1);
  convN(bufA, EXT + 2,  S2,   256, 19, 19, 512, 10, 10, 3, 2, 1, 1);    // source2
  convN(S2,   EXT + 4,  bufA, 512, 10, 10, 128, 10, 10, 1, 1, 0, 1);
  convN(bufA, EXT + 6,  S3,   128, 10, 10, 256, 5,  5,  3, 2, 1, 1);    // source3
  convN(S3,   EXT + 8,  bufA, 256, 5,  5,  128, 5,  5,  1, 1, 0, 1);
  convN(bufA, EXT + 10, S4,   128, 5,  5,  256, 3,  3,  3, 1, 0, 1);    // source4
  convN(S4,   EXT + 12, bufA, 256, 3,  3,  128, 3,  3,  1, 1, 0, 1);
  convN(bufA, EXT + 14, S5,   128, 3,  3,  256, 1,  1,  3, 1, 0, 1);    // source5

  // --- detection heads: write channel-last f32 directly into the concatenated output ---
  float* outF = (float*)d_out;
  const int LOCSTRIDE = 8732 * 4, CONFSTRIDE = 8732 * 21;
  float* confOut = outF + 4 * LOCSTRIDE;
  const int offs[6] = {0, 5776, 7942, 8542, 8692, 8728};
  const _Float16* srcs[6] = {S0, S1, S2, S3, S4, S5};
  const int sch[6] = {512, 1024, 512, 256, 256, 256};
  const int sHW[6] = {38, 19, 10, 5, 3, 1};
  const int asp[6] = {4, 6, 6, 6, 4, 4};
  for (int i = 0; i < 6; ++i) {
    convHead(srcs[i], LOC  + 2 * i, outF,    sch[i], sHW[i], sHW[i], asp[i] * 4,  LOCSTRIDE,  offs[i] * 4);
    convHead(srcs[i], CONF + 2 * i, confOut, sch[i], sHW[i], sHW[i], asp[i] * 21, CONFSTRIDE, offs[i] * 21);
  }

  // --- dbox passthrough ---
  float* dboxOut = outF + 4 * LOCSTRIDE + 4 * CONFSTRIDE;
  copy_kernel<<<(8732 * 4 + 255) / 256, 256, 0, stream>>>(dbox, dboxOut, 8732 * 4);
}
